// MultiHeadAttention_33973191311882
// MI455X (gfx1250) — matrix operand order repaired and checked by reading, not compiled, not run on hardware
//
#include <hip/hip_runtime.h>

// ---------------------------------------------------------------------------
// MHA forward for MI455X (gfx1250, wave32, WMMA + Tensor Data Mover).
// Pipeline: f32->bf16 convert -> 3x GEMM (Q/K/V) -> RoPE -> V transpose ->
//           flash-attn -> GEMM (O@Wo^T -> f32 out).
// Matrix math: v_wmma_f32_16x16x32_bf16.  Tile staging: TDM tensor_load_to_lds
// (double-buffered, TENSORcnt-waited) when the builtin exists, vec8 copies
// otherwise.  All LDS tiles use padded row strides produced directly by the
// TDM pad_interval/pad_amount fields.
// ---------------------------------------------------------------------------

#define B_  2
#define S_  2048
#define D_  2048
#define H_  16
#define HD_ 128

typedef __attribute__((ext_vector_type(16))) __bf16 v16bf;
typedef __attribute__((ext_vector_type(8)))  __bf16 v8bf;
typedef __attribute__((ext_vector_type(4)))  __bf16 v4bf;
typedef __attribute__((ext_vector_type(8)))  float  v8f;
typedef __attribute__((ext_vector_type(4)))  unsigned int u32x4;
typedef __attribute__((ext_vector_type(4)))  int i32x4;
typedef __attribute__((ext_vector_type(8)))  int i32x8;

#if __has_builtin(__builtin_amdgcn_tensor_load_to_lds) && \
    __has_builtin(__builtin_amdgcn_s_wait_tensorcnt)
#define USE_TDM 1
#else
#define USE_TDM 0
#endif

// Build a 16-element bf16 fragment from two contiguous 16B LDS/global loads.
__device__ __forceinline__ v16bf make_frag(const __bf16* lo, const __bf16* hi) {
    v8bf a = *(const v8bf*)lo;
    v8bf b = *(const v8bf*)hi;
    return __builtin_shufflevector(a, b, 0,1,2,3,4,5,6,7,8,9,10,11,12,13,14,15);
}

__device__ __forceinline__ v8f wmma_bf16(v16bf a, v16bf b, v8f c) {
    return __builtin_amdgcn_wmma_f32_16x16x32_bf16(false, a, false, b, (short)0, c,
                                                   false, false);
}

#if USE_TDM
// Issue a 2D TENSOR_LOAD_TO_LDS.  d0/d1 = tile (== tensor) dims in bf16
// elements, stride0 = row stride in elements, lds_off = byte offset in the
// workgroup's LDS.  pad_interval: pad after 2^(v+1) DWORDs; pad_amount:
// (v+1) DWORDs of padding -> reproduces our padded LDS row strides.
__device__ __forceinline__ void tdm_load_2d(unsigned lds_off, const void* gsrc,
                                            unsigned d0, unsigned d1,
                                            unsigned stride0,
                                            unsigned pad_interval,
                                            unsigned pad_amount) {
    unsigned long long ga = (unsigned long long)(size_t)gsrc;
    u32x4 g0;
    g0[0] = 1u;                                   // count=1 (valid user D#)
    g0[1] = lds_off;                              // lds_addr (bytes)
    g0[2] = (unsigned)ga;                         // global_addr[31:0]
    g0[3] = (unsigned)((ga >> 32) & 0x1FFFFFFull) // global_addr[56:32]
          | (2u << 30);                           // type = 2 ("image")
    i32x8 g1;
    g1[0] = (int)((1u << 16)                      // data_size = 1 (2 bytes)
          | (1u << 20)                            // pad_enable
          | (pad_interval << 22) | (pad_amount << 25));
    g1[1] = (int)((d0 & 0xFFFFu) << 16);          // tensor_dim0[15:0]
    g1[2] = (int)((d0 >> 16) | ((d1 & 0xFFFFu) << 16));   // dim0 hi | dim1 lo
    g1[3] = (int)((d1 >> 16) | (d0 << 16));       // dim1 hi | tile_dim0
    g1[4] = (int)(d1 & 0xFFFFu);                  // tile_dim1 (tile_dim2 = 0)
    g1[5] = (int)stride0;                         // tensor_dim0_stride[31:0]
    g1[6] = 0;
    g1[7] = 0;                                    // dim1_stride = 0 (2D)
    i32x4 z4 = {0, 0, 0, 0};
#if __clang_major__ >= 23
    i32x8 z8 = {0, 0, 0, 0, 0, 0, 0, 0};
    __builtin_amdgcn_tensor_load_to_lds(g0, g1, z4, z4, z8, 0);
#else
    __builtin_amdgcn_tensor_load_to_lds(g0, g1, z4, z4, 0);
#endif
}
#endif

// ---------------------------------------------------------------------------
// f32 -> bf16 bulk convert (vec4).
// ---------------------------------------------------------------------------
__global__ __launch_bounds__(256)
void cvt_bf16(const float* __restrict__ src, __bf16* __restrict__ dst, int n4) {
    int i = blockIdx.x * 256 + threadIdx.x;
    if (i < n4) {
        float4 f = ((const float4*)src)[i];
        ((v4bf*)dst)[i] = v4bf{(__bf16)f.x, (__bf16)f.y, (__bf16)f.z, (__bf16)f.w};
    }
}

// ---------------------------------------------------------------------------
// C[M,N] = A[M,K] @ W[N,K]^T, all-bf16 operands, f32 accumulate.
// 128x128 tile, K-step 32, 256 threads (8 waves, 4x2; each wave = 2x4 WMMA
// tiles).  Double-buffered LDS (rows padded to 40 halves), staged by TDM.
// ---------------------------------------------------------------------------
template<typename TC>
__global__ __launch_bounds__(256)
void gemm_bt(const __bf16* __restrict__ A, const __bf16* __restrict__ W,
             TC* __restrict__ C, int M, int N, int K) {
    extern __shared__ __align__(16) char smem[];   // As@0, Bs@10240 (x2 bufs)
    const int tid  = threadIdx.x;
    const int lane = tid & 31;
    const int w    = tid >> 5;
    const int wm   = w >> 1, wn = w & 1;
    const int m0   = blockIdx.y * 128, n0 = blockIdx.x * 128;
    const int r    = lane & 15, hif = lane >> 4;

    const v8f vz = {0.f,0.f,0.f,0.f,0.f,0.f,0.f,0.f};
    v8f acc[2][4];
    #pragma unroll
    for (int mt = 0; mt < 2; ++mt)
        #pragma unroll
        for (int nt = 0; nt < 4; ++nt) acc[mt][nt] = vz;

    const int KI = K / 32;
    int cur = 0;

#if USE_TDM
    if (w == 0) {   // prologue: DMA first K-slab into buffer 0
        tdm_load_2d(0,     A + (size_t)m0 * K, 32, 128, (unsigned)K, 3, 3);
        tdm_load_2d(10240, W + (size_t)n0 * K, 32, 128, (unsigned)K, 3, 3);
        __builtin_amdgcn_s_wait_tensorcnt((short)0);
    }
    __syncthreads();
#endif

    for (int it = 0; it < KI; ++it) {
        __bf16* As = (__bf16*)(smem + cur * 20480);
        __bf16* Bs = (__bf16*)(smem + cur * 20480 + 10240);
#if USE_TDM
        if (w == 0 && it + 1 < KI) {       // DMA next slab into other buffer
            int kn = (it + 1) * 32;
            unsigned o = (unsigned)((cur ^ 1) * 20480);
            tdm_load_2d(o,         A + (size_t)m0 * K + kn, 32, 128, (unsigned)K, 3, 3);
            tdm_load_2d(o + 10240, W + (size_t)n0 * K + kn, 32, 128, (unsigned)K, 3, 3);
        }
#else
        __syncthreads();
        {   // synchronous vec8 staging into buffer 0
            int kc = it * 32;
            #pragma unroll
            for (int i = 0; i < 2; ++i) {
                int v = tid + i * 256;              // 512 vec8 slots
                int row = v >> 2, c8 = (v & 3) * 8;
                *(v8bf*)&As[row * 40 + c8] =
                    *(const v8bf*)(A + (size_t)(m0 + row) * K + kc + c8);
                *(v8bf*)&Bs[row * 40 + c8] =
                    *(const v8bf*)(W + (size_t)(n0 + row) * K + kc + c8);
            }
        }
        __syncthreads();
#endif
        // Fragments: A and B lane-half -> K{0-7,16-23}|{8-15,24-31}.
        v16bf af[2], bfr[4];
        const int kbA = hif * 8, kbB = hif * 8;
        #pragma unroll
        for (int mt = 0; mt < 2; ++mt) {
            const __bf16* ap = As + (wm * 32 + mt * 16 + r) * 40;
            af[mt] = make_frag(ap + kbA, ap + kbA + 16);
        }
        #pragma unroll
        for (int nt = 0; nt < 4; ++nt) {
            const __bf16* bp = Bs + (wn * 64 + nt * 16 + r) * 40;
            bfr[nt] = make_frag(bp + kbB, bp + kbB + 16);
        }
        #pragma unroll
        for (int mt = 0; mt < 2; ++mt)
            #pragma unroll
            for (int nt = 0; nt < 4; ++nt)
                acc[mt][nt] = wmma_bf16(af[mt], bfr[nt], acc[mt][nt]);
#if USE_TDM
        if (w == 0 && it + 1 < KI) __builtin_amdgcn_s_wait_tensorcnt((short)0);
        __syncthreads();
        cur ^= 1;
#endif
    }

    // C layout: VGPR rr -> rows (rr | rr+8) per lane half, col = lane&15.
    #pragma unroll
    for (int mt = 0; mt < 2; ++mt) {
        #pragma unroll
        for (int nt = 0; nt < 4; ++nt) {
            int col = n0 + wn * 64 + nt * 16 + r;
            #pragma unroll
            for (int rr = 0; rr < 8; ++rr) {
                int row = m0 + wm * 32 + mt * 16 + rr + hif * 8;
                C[(size_t)row * N + col] = (TC)acc[mt][nt][rr];
            }
        }
    }
}

// ---------------------------------------------------------------------------
// RoPE in-place on bf16 Q and K.  One thread per rotation pair.
// ---------------------------------------------------------------------------
__global__ __launch_bounds__(256)
void rope_kernel(__bf16* __restrict__ Q, __bf16* __restrict__ Kk) {
    int t  = blockIdx.x * blockDim.x + threadIdx.x;
    int i  = t & 63;
    int h  = (t >> 6) & (H_ - 1);
    int bs = t >> 10;                               // b*S + s
    int s  = bs & (S_ - 1);
    float inv = __expf(-((float)(2 * i) * (1.0f / 128.f)) * 9.210340371976184f);
    float ang = (float)s * inv;
    float sn, cs;
    __sincosf(ang, &sn, &cs);
    size_t base = (size_t)bs * D_ + (size_t)h * HD_ + i;
    float q1 = (float)Q[base], q2 = (float)Q[base + 64];
    Q[base]      = (__bf16)(q1 * cs - q2 * sn);
    Q[base + 64] = (__bf16)(q1 * sn + q2 * cs);
    float k1 = (float)Kk[base], k2 = (float)Kk[base + 64];
    Kk[base]      = (__bf16)(k1 * cs - k2 * sn);
    Kk[base + 64] = (__bf16)(k1 * sn + k2 * cs);
}

// ---------------------------------------------------------------------------
// V (b,s,h,hd) -> Vt (b,h,hd,s): one global transpose so attention V tiles
// become contiguous row loads (kills the LDS byte-scatter transpose).
// ---------------------------------------------------------------------------
__global__ __launch_bounds__(256)
void transpose_v(const __bf16* __restrict__ V, __bf16* __restrict__ Vt) {
    __shared__ __bf16 t[32][34];
    int b  = blockIdx.z;
    int s0 = blockIdx.x * 32, d0 = blockIdx.y * 32;   // d never straddles a head
    int tx = threadIdx.x & 31, ty4 = (threadIdx.x >> 5) * 4;
    #pragma unroll
    for (int i = 0; i < 4; ++i)
        t[ty4 + i][tx] = V[((size_t)b * S_ + s0 + ty4 + i) * D_ + d0 + tx];
    __syncthreads();
    int h = d0 >> 7, hd0 = d0 & 127;
    #pragma unroll
    for (int i = 0; i < 4; ++i) {
        int dy = ty4 + i;
        Vt[(((size_t)(b * H_ + h) * HD_) + hd0 + dy) * S_ + s0 + tx] = t[tx][dy];
    }
}

// ---------------------------------------------------------------------------
// Causal flash attention.  Grid (S/64, B*H), 128 threads = 4 waves x 16 q rows.
// Double-buffered K[64][136] + Vt[128][72] tiles (TDM-staged), per-wave P tile,
// online softmax via wave32 xor-shuffles.
// ---------------------------------------------------------------------------
__global__ __launch_bounds__(128)
void attn_kernel(const __bf16* __restrict__ Q, const __bf16* __restrict__ Kb,
                 const __bf16* __restrict__ Vtg, __bf16* __restrict__ O) {
    extern __shared__ __align__(16) char smem[];
    constexpr unsigned KS_SZ  = 64 * 136 * 2;        // 17408
    constexpr unsigned VT_SZ  = 128 * 72 * 2;        // 18432
    constexpr unsigned BUF_SZ = KS_SZ + VT_SZ;       // 35840
    constexpr unsigned PW_OFF = 2 * BUF_SZ;          // 71680

    const int tid  = threadIdx.x;
    const int lane = tid & 31;
    const int w    = tid >> 5;
    const int qb   = blockIdx.x;
    const int bh   = blockIdx.y;
    const int b    = bh >> 4, h = bh & (H_ - 1);
    const int q0   = qb * 64 + w * 16;
    const int r    = lane & 15, hif = lane >> 4;
    const size_t rowbase = (size_t)b * S_ * D_ + (size_t)h * HD_;
    const __bf16* Vtg_bh = Vtg + (size_t)(b * H_ + h) * HD_ * S_;
    const float scale = 0.08838834764831845f;        // 1/sqrt(128)

    // Preload this wave's Q fragments (16 rows x 128 hd).
    v16bf qf[4];
    {
        const __bf16* qrow = Q + rowbase + (size_t)(q0 + r) * D_;
        #pragma unroll
        for (int kc = 0; kc < 4; ++kc) {
            int kb = kc * 32 + hif * 8;
            qf[kc] = make_frag(qrow + kb, qrow + kb + 16);
        }
    }

    const v8f vz = {0.f,0.f,0.f,0.f,0.f,0.f,0.f,0.f};
    v8f oacc[8];
    #pragma unroll
    for (int i = 0; i < 8; ++i) oacc[i] = vz;
    float mrow[8], lrow[8];
    #pragma unroll
    for (int i = 0; i < 8; ++i) { mrow[i] = -1e30f; lrow[i] = 0.f; }

    auto stage = [&](int j, int buf) {
#if USE_TDM
        if (w == 0) {
            tdm_load_2d(buf * BUF_SZ, Kb + rowbase + (size_t)(j * 64) * D_,
                        128, 64, D_, 5, 3);                 // -> stride 136
            tdm_load_2d(buf * BUF_SZ + KS_SZ, Vtg_bh + (size_t)(j * 64),
                        64, 128, S_, 4, 3);                 // -> stride 72
        }
#else
        __bf16* Ks = (__bf16*)(smem + buf * BUF_SZ);
        __bf16* Vt = (__bf16*)(smem + buf * BUF_SZ + KS_SZ);
        #pragma unroll
        for (int i = 0; i < 8; ++i) {
            int v   = tid + i * 128;
            int key = v >> 4, c8 = (v & 15) * 8;
            *(v8bf*)&Ks[key * 136 + c8] =
                *(const v8bf*)(Kb + rowbase + (size_t)(j * 64 + key) * D_ + c8);
            int hd = v >> 3, k8 = (v & 7) * 8;
            *(v8bf*)&Vt[hd * 72 + k8] =
                *(const v8bf*)(Vtg_bh + (size_t)hd * S_ + j * 64 + k8);
        }
#endif
    };

    int cur = 0;
    const int nkb = qb + 1;
#if USE_TDM
    stage(0, 0);
    if (w == 0) __builtin_amdgcn_s_wait_tensorcnt((short)0);
    __syncthreads();
#endif
    for (int j = 0; j < nkb; ++j) {
#if USE_TDM
        if (j + 1 < nkb) stage(j + 1, cur ^ 1);
#else
        __syncthreads();
        stage(j, 0);
        __syncthreads();
#endif
        const __bf16* Ks = (const __bf16*)(smem + cur * BUF_SZ);
        const __bf16* Vt = (const __bf16*)(smem + cur * BUF_SZ + KS_SZ);
        __bf16*       Pw = (__bf16*)(smem + PW_OFF);

        // Scores: S = Q @ K^T  (16 q x 64 keys).
        v8f sc[4];
        #pragma unroll
        for (int nt = 0; nt < 4; ++nt) sc[nt] = vz;
        #pragma unroll
        for (int kc = 0; kc < 4; ++kc) {
            #pragma unroll
            for (int nt = 0; nt < 4; ++nt) {
                const __bf16* kp = Ks + (nt * 16 + r) * 136 + kc * 32 + hif * 8;
                sc[nt] = wmma_bf16(qf[kc], make_frag(kp, kp + 16), sc[nt]);
            }
        }

        // Scale + causal mask (analytic).
        #pragma unroll
        for (int nt = 0; nt < 4; ++nt) {
            int col = j * 64 + nt * 16 + r;
            #pragma unroll
            for (int rr = 0; rr < 8; ++rr) {
                int qr = q0 + rr + hif * 8;
                float x = sc[nt][rr] * scale;
                sc[nt][rr] = (col <= qr) ? x : -1e30f;
            }
        }

        // Online softmax (row lives in 16 lanes of one half).
        #pragma unroll
        for (int rr = 0; rr < 8; ++rr) {
            float cm = fmaxf(fmaxf(sc[0][rr], sc[1][rr]),
                             fmaxf(sc[2][rr], sc[3][rr]));
            cm = fmaxf(cm, __shfl_xor(cm, 1));
            cm = fmaxf(cm, __shfl_xor(cm, 2));
            cm = fmaxf(cm, __shfl_xor(cm, 4));
            cm = fmaxf(cm, __shfl_xor(cm, 8));
            float mn = fmaxf(mrow[rr], cm);
            float f  = __expf(mrow[rr] - mn);
            mrow[rr] = mn;
            float ss = 0.f;
            #pragma unroll
            for (int nt = 0; nt < 4; ++nt) {
                float p = __expf(sc[nt][rr] - mn);
                sc[nt][rr] = p;
                ss += p;
            }
            ss += __shfl_xor(ss, 1);
            ss += __shfl_xor(ss, 2);
            ss += __shfl_xor(ss, 4);
            ss += __shfl_xor(ss, 8);
            lrow[rr] = lrow[rr] * f + ss;
            #pragma unroll
            for (int nt = 0; nt < 8; ++nt) oacc[nt][rr] *= f;
        }

        // Spill P (bf16) to this wave's private LDS tile.
        #pragma unroll
        for (int nt = 0; nt < 4; ++nt)
            #pragma unroll
            for (int rr = 0; rr < 8; ++rr)
                Pw[(w * 16 + rr + hif * 8) * 72 + nt * 16 + r] = (__bf16)sc[nt][rr];
        asm volatile("s_wait_dscnt 0" ::: "memory");

        // O += P @ V  (16x64)@(64x128).
        #pragma unroll
        for (int ks = 0; ks < 2; ++ks) {
            const __bf16* pp = Pw + (w * 16 + r) * 72 + ks * 32 + hif * 8;
            v16bf pf = make_frag(pp, pp + 16);
            #pragma unroll
            for (int nt = 0; nt < 8; ++nt) {
                const __bf16* vp = Vt + (nt * 16 + r) * 72 + ks * 32 + hif * 8;
                oacc[nt] = wmma_bf16(pf, make_frag(vp, vp + 16), oacc[nt]);
            }
        }
#if USE_TDM
        if (w == 0 && j + 1 < nkb) __builtin_amdgcn_s_wait_tensorcnt((short)0);
        __syncthreads();
        cur ^= 1;
#endif
    }

    // Normalize and store bf16 output at (b, q, h*128 + hd).
    float invl[8];
    #pragma unroll
    for (int rr = 0; rr < 8; ++rr) invl[rr] = 1.0f / lrow[rr];
    #pragma unroll
    for (int nt = 0; nt < 8; ++nt) {
        int col = nt * 16 + r;
        #pragma unroll
        for (int rr = 0; rr < 8; ++rr) {
            int qr = q0 + rr + hif * 8;
            O[rowbase + (size_t)qr * D_ + col] = (__bf16)(oacc[nt][rr] * invl[rr]);
        }
    }
}

// ---------------------------------------------------------------------------
// Harness entry.  Inputs: x(f32), mask(unused; causal is analytic), Wq, Wk,
// Wv, Wo (f32).  Workspace (~118 MB): xb, Wq/Wk/Wv/Wo bf16, Q, K, V, Vt;
// attention output aliases the dead V buffer.
// ---------------------------------------------------------------------------
extern "C" void kernel_launch(void* const* d_in, const int* in_sizes, int n_in,
                              void* d_out, int out_size, void* d_ws, size_t ws_size,
                              hipStream_t stream) {
    (void)in_sizes; (void)n_in; (void)out_size; (void)ws_size;
    const float* x  = (const float*)d_in[0];
    const float* Wq = (const float*)d_in[2];
    const float* Wk = (const float*)d_in[3];
    const float* Wv = (const float*)d_in[4];
    const float* Wo = (const float*)d_in[5];
    float* out = (float*)d_out;

    const size_t nel = (size_t)B_ * S_ * D_;    // 8,388,608
    const size_t dd  = (size_t)D_ * D_;         // 4,194,304
    __bf16* xb  = (__bf16*)d_ws;
    __bf16* Wqb = xb  + nel;
    __bf16* Wkb = Wqb + dd;
    __bf16* Wvb = Wkb + dd;
    __bf16* Wob = Wvb + dd;
    __bf16* Qb  = Wob + dd;
    __bf16* Kb  = Qb  + nel;
    __bf16* Vb  = Kb  + nel;
    __bf16* Vtg = Vb  + nel;
    __bf16* Ob  = Vb;                           // reuse: V dead after transpose

    cvt_bf16<<<(int)(nel / 4 / 256), 256, 0, stream>>>(x,  xb,  (int)(nel / 4));
    cvt_bf16<<<(int)(dd  / 4 / 256), 256, 0, stream>>>(Wq, Wqb, (int)(dd / 4));
    cvt_bf16<<<(int)(dd  / 4 / 256), 256, 0, stream>>>(Wk, Wkb, (int)(dd / 4));
    cvt_bf16<<<(int)(dd  / 4 / 256), 256, 0, stream>>>(Wv, Wvb, (int)(dd / 4));
    cvt_bf16<<<(int)(dd  / 4 / 256), 256, 0, stream>>>(Wo, Wob, (int)(dd / 4));

    dim3 gg(D_ / 128, (B_ * S_) / 128);         // (16, 32)
    gemm_bt<__bf16><<<gg, 256, 40960, stream>>>(xb, Wqb, Qb, B_ * S_, D_, D_);
    gemm_bt<__bf16><<<gg, 256, 40960, stream>>>(xb, Wkb, Kb, B_ * S_, D_, D_);
    gemm_bt<__bf16><<<gg, 256, 40960, stream>>>(xb, Wvb, Vb, B_ * S_, D_, D_);

    rope_kernel<<<(B_ * S_ * H_ * 64) / 256, 256, 0, stream>>>(Qb, Kb);

    dim3 gt(S_ / 32, D_ / 32, B_);              // (64, 64, 2)
    transpose_v<<<gt, 256, 0, stream>>>(Vb, Vtg);

    dim3 ga(S_ / 64, B_ * H_);                  // (32, 32)
    attn_kernel<<<ga, 128, 80896, stream>>>(Qb, Kb, Vtg, Ob);

    gemm_bt<float><<<gg, 256, 40960, stream>>>(Ob, Wob, out, B_ * S_, D_, D_);
}
